// PackedBitNetMNIST_57939108823186
// MI455X (gfx1250) — matrix operand order repaired and checked by reading, not compiled, not run on hardware
//
#include <hip/hip_runtime.h>
#include <hip/hip_bf16.h>

// -------------------------------------------------------------------------
// PackedBitNetMNIST on MI455X (gfx1250, wave32, WMMA + async-to-LDS)
//
// h = gelu(rmsnorm(x @ W1^T)); h = gelu(rmsnorm(h @ W2^T)); out = h @ W3^T
// Wk are 2-bit packed ternary (code {0,1,2} -> {-1,0,+1} * scale).
// The per-row activation scale in the reference cancels algebraically, so
// each bitlinear is a plain GEMM against the dequantized ternary weight.
// Matrix math: v_wmma_f32_16x16x32_bf16 (f32 accumulate).
// A tiles staged with global_load_async_to_lds_b128 (ASYNCcnt), packed
// weights unpacked in-register to bf16 {-s,0,+s} into LDS.
// Layer 3 (N=10) is K-split 8x over blockIdx.z with a deterministic
// fixed-order partial reduction (no float atomics).
// -------------------------------------------------------------------------

#define BATCH   16384
#define HIDDEN  4096
#define IN_F    784
#define IN_PAD  800      // 784 padded to multiple of 32
#define NCLASS  10
#define KSPLIT  8        // k-chunks for the classifier layer
#define NPAD    16       // padded partial width

typedef unsigned short u16;
typedef unsigned int   u32;

typedef __attribute__((ext_vector_type(16))) __bf16 bf16x16;
typedef __attribute__((ext_vector_type(8)))  __bf16 bf16x8;
typedef __attribute__((ext_vector_type(8)))  float  f32x8;
typedef __attribute__((ext_vector_type(8)))  u16    u16x8;

typedef __attribute__((address_space(3))) const void lds_cv;

__device__ __forceinline__ u16 f32_to_bf16_rne(float f) {
  u32 u = __builtin_bit_cast(u32, f);
  u32 r = u + 0x7FFFu + ((u >> 16) & 1u);   // round to nearest even
  return (u16)(r >> 16);
}
__device__ __forceinline__ float bf16_to_f32(u16 h) {
  return __builtin_bit_cast(float, (u32)h << 16);
}

// byte offset of a shared-memory pointer within the workgroup LDS allocation
__device__ __forceinline__ u32 lds_off(const void* p) {
  return (u32)(size_t)(lds_cv*)p;
}

// async DMA: 16 bytes global -> LDS, tracked by ASYNCcnt (GVS addressing)
__device__ __forceinline__ void async_ld_b128(u32 ldsoff, u32 goff,
                                              const void* base) {
  asm volatile("global_load_async_to_lds_b128 %0, %1, %2"
               :: "v"(ldsoff), "v"(goff), "s"(base) : "memory");
}
__device__ __forceinline__ void wait_asynccnt0() {
  asm volatile("s_wait_asynccnt 0x0" ::: "memory");
}

// Load a 16-element bf16 fragment from 32 contiguous LDS bytes.
__device__ __forceinline__ bf16x16 load_frag16(const u16* p) {
  u16x8 lo = *(const u16x8*)p;
  u16x8 hi = *(const u16x8*)(p + 8);
  bf16x8 l = __builtin_bit_cast(bf16x8, lo);
  bf16x8 h = __builtin_bit_cast(bf16x8, hi);
  return __builtin_shufflevector(l, h, 0,1,2,3,4,5,6,7,8,9,10,11,12,13,14,15);
}

// -------------------------------------------------------------------------
// GEMM: C[M,N] = A[M,K] (bf16, row major, lda = padded K) x W^T
// W packed ternary: Wp[N][kwords] int32, 16 x 2-bit codes per word (LSB 1st)
// Block tile 128x256x32, 8 waves (2x4), wave tile 64x64 -> 4x4 WMMA accums.
// blockIdx.z selects a K-chunk of `ksteps` tiles (c_zstride-strided output).
// -------------------------------------------------------------------------
#define BM 128
#define BN 256
#define BK 32
#define APAD 40   // row pitch in u16 (80 B) -> conflict-free b128 frag reads

template<bool BF16OUT>
__global__ __launch_bounds__(256)
void bitnet_gemm_wmma(const u16* __restrict__ A, int lda,
                      const int* __restrict__ Wp, int kwords, int ksteps,
                      const float* __restrict__ scale, int N, int Nout,
                      float* __restrict__ Cf, u16* __restrict__ Cb, int ldc,
                      size_t c_zstride)
{
  __shared__ u16 As[2][BM][APAD];
  __shared__ u16 Bs[2][BN][APAD];

  const int t    = threadIdx.x;
  const int lane = t & 31;
  const int wid  = t >> 5;
  const int wm   = wid >> 2;           // 0..1  (64 rows each)
  const int wn   = wid & 3;            // 0..3  (64 cols each)
  const int lm   = lane & 15;
  const int sel  = lane >> 4;          // fragment half-select

  const int m0   = blockIdx.y * BM;
  const int n0   = blockIdx.x * BN;
  const int koff = blockIdx.z * ksteps;      // first k-tile of this chunk

  // ternary dequant table in bf16: code 0 -> -s, 1 -> 0, 2 -> +s
  const float s = scale[0];
  const u16 tp = f32_to_bf16_rne(s);
  const u16 tab[4] = { (u16)(tp ^ 0x8000u), (u16)0, tp, tp };

  const int ldv = lda >> 3;            // row pitch in 16-byte units

  // A tile = 128 rows x 4 chunks(16B); 512 chunks, 2 per thread.
  // LDS columns hold k in natural order [0..31]
  const int c0r = t >> 2,          c0g = t & 3;
  const int c1r = (t + 256) >> 2,  c1g = (t + 256) & 3;
  const int c0p = c0g;
  const int c1p = c1g;

  // --- async-stage A k-tile `step` (absolute) into LDS buffer `buf` ---
  auto a_stage = [&](int step, int buf) {
    const u32 kb = (u32)(step * (BK / 8));     // k-tile offset, 16B units
    async_ld_b128(lds_off(&As[buf][c0r][c0p * 8]),
                  (u32)(((size_t)(m0 + c0r) * ldv + kb + c0g) * 16), A);
    async_ld_b128(lds_off(&As[buf][c1r][c1p * 8]),
                  (u32)(((size_t)(m0 + c1r) * ldv + kb + c1g) * 16), A);
  };

  // --- B packed-word fetch (512 words/tile, 2 per thread) ------------
  int bw0, bw1;
  const int bn0 = n0 + (t >> 1);
  const int bn1 = n0 + 128 + (t >> 1);
  auto b_fetch = [&](int step) {
    const int widx = step * 2 + (t & 1);
    const bool kv = (widx < kwords);
    // 0x55555555 = all code-1 = weight 0 (matches reference zero padding)
    bw0 = (kv && bn0 < N) ? Wp[(size_t)bn0 * kwords + widx] : (int)0x55555555;
    bw1 = (kv && bn1 < N) ? Wp[(size_t)bn1 * kwords + widx] : (int)0x55555555;
  };
  auto b_unpack = [&](int buf) {
    u32* d0 = (u32*)&Bs[buf][t >> 1][(t & 1) * 16];
    u32* d1 = (u32*)&Bs[buf][128 + (t >> 1)][(t & 1) * 16];
    #pragma unroll
    for (int jj = 0; jj < 8; ++jj) {
      u32 a0 = ((u32)bw0 >> (4 * jj)) & 3u, a1 = ((u32)bw0 >> (4 * jj + 2)) & 3u;
      u32 b0 = ((u32)bw1 >> (4 * jj)) & 3u, b1 = ((u32)bw1 >> (4 * jj + 2)) & 3u;
      d0[jj] = (u32)tab[a0] | ((u32)tab[a1] << 16);
      d1[jj] = (u32)tab[b0] | ((u32)tab[b1] << 16);
    }
  };

  f32x8 acc[4][4];
  const f32x8 fz = {0.f,0.f,0.f,0.f,0.f,0.f,0.f,0.f};
  #pragma unroll
  for (int fm = 0; fm < 4; ++fm)
    #pragma unroll
    for (int fn = 0; fn < 4; ++fn) acc[fm][fn] = fz;

  // prologue: first tile of this chunk into buffer 0
  a_stage(koff, 0);
  b_fetch(koff);
  b_unpack(0);
  wait_asynccnt0();
  __syncthreads();

  for (int stp = 0; stp < ksteps; ++stp) {
    const int  cur  = stp & 1;
    const bool more = (stp + 1 < ksteps);
    if (more) {                         // overlap next tile with compute
      a_stage(koff + stp + 1, cur ^ 1);
      b_fetch(koff + stp + 1);
    }

    bf16x16 bfr[4];
    #pragma unroll
    for (int fn = 0; fn < 4; ++fn)
      bfr[fn] = load_frag16(&Bs[cur][wn * 64 + fn * 16 + lm][sel * 16]);
    #pragma unroll
    for (int fm = 0; fm < 4; ++fm) {
      bf16x16 af = load_frag16(&As[cur][wm * 64 + fm * 16 + lm][sel * 16]);
      #pragma unroll
      for (int fn = 0; fn < 4; ++fn)
        acc[fm][fn] = __builtin_amdgcn_wmma_f32_16x16x32_bf16(
            false, af, false, bfr[fn], (short)0, acc[fm][fn], false, false);
    }

    if (more) {
      b_unpack(cur ^ 1);
      wait_asynccnt0();                 // our async A writes landed
      __syncthreads();                  // everyone's tile s+1 visible
    }
  }

  // --- writeback: D layout lane=(n), vgpr j -> m = j + 8*sel ----------
  float* Cfz = Cf + (size_t)blockIdx.z * c_zstride;
  #pragma unroll
  for (int fm = 0; fm < 4; ++fm) {
    #pragma unroll
    for (int fn = 0; fn < 4; ++fn) {
      const int mrow = m0 + wm * 64 + fm * 16 + sel * 8;
      const int ncol = n0 + wn * 64 + fn * 16 + lm;
      if (ncol < Nout) {
        #pragma unroll
        for (int j = 0; j < 8; ++j) {
          const size_t off = (size_t)(mrow + j) * ldc + ncol;
          if constexpr (BF16OUT) Cb[off] = f32_to_bf16_rne(acc[fm][fn][j]);
          else                   Cfz[off] = acc[fm][fn][j];
        }
      }
    }
  }
}

// -------------------------------------------------------------------------
// fp32 x [B,784] -> bf16 [B,800] (zero padded K)
// -------------------------------------------------------------------------
__global__ __launch_bounds__(256)
void convert_pad_kernel(const float* __restrict__ x, u16* __restrict__ bx)
{
  const int row = blockIdx.x;
  for (int c = threadIdx.x; c < IN_PAD; c += 256)
    bx[(size_t)row * IN_PAD + c] =
        (c < IN_F) ? f32_to_bf16_rne(x[(size_t)row * IN_F + c]) : (u16)0;
}

// -------------------------------------------------------------------------
// In-place rmsnorm + exact-erf gelu on bf16 rows of width 4096.
// One 256-thread block per row, 16 elements/thread.
// -------------------------------------------------------------------------
__global__ __launch_bounds__(256)
void rmsnorm_gelu_kernel(u16* __restrict__ h, const float* __restrict__ gamma)
{
  const int row = blockIdx.x;
  const int t   = threadIdx.x;
  const size_t base = (size_t)row * HIDDEN + (size_t)t * 16;

  u16x8 r0 = *(const u16x8*)(h + base);
  u16x8 r1 = *(const u16x8*)(h + base + 8);

  float v[16];
  float ss = 0.f;
  #pragma unroll
  for (int i = 0; i < 8; ++i) { v[i]     = bf16_to_f32(r0[i]); }
  #pragma unroll
  for (int i = 0; i < 8; ++i) { v[i + 8] = bf16_to_f32(r1[i]); }
  #pragma unroll
  for (int i = 0; i < 16; ++i) ss += v[i] * v[i];

  __shared__ float red[256];
  red[t] = ss;
  __syncthreads();
  #pragma unroll
  for (int off = 128; off > 0; off >>= 1) {
    if (t < off) red[t] += red[t + off];
    __syncthreads();
  }
  const float rinv = rsqrtf(red[0] * (1.0f / HIDDEN) + 1e-6f);

  u16x8 o0, o1;
  #pragma unroll
  for (int i = 0; i < 16; ++i) {
    float y = v[i] * rinv * gamma[t * 16 + i];
    y = 0.5f * y * (1.0f + erff(y * 0.70710678118654752f));  // exact gelu
    u16 b = f32_to_bf16_rne(y);
    if (i < 8) o0[i] = b; else o1[i - 8] = b;
  }
  *(u16x8*)(h + base)     = o0;
  *(u16x8*)(h + base + 8) = o1;
}

// -------------------------------------------------------------------------
// Deterministic fixed-order sum of KSPLIT partials: [z][B][16] -> [B][10]
// -------------------------------------------------------------------------
__global__ __launch_bounds__(256)
void ksplit_reduce_kernel(const float* __restrict__ part,
                          float* __restrict__ out)
{
  const int idx = blockIdx.x * 256 + threadIdx.x;
  if (idx >= BATCH * NCLASS) return;
  const int m = idx / NCLASS;
  const int n = idx - m * NCLASS;
  float s = 0.f;
  #pragma unroll
  for (int z = 0; z < KSPLIT; ++z)
    s += part[(size_t)z * BATCH * NPAD + (size_t)m * NPAD + n];
  out[(size_t)m * NCLASS + n] = s;
}

// -------------------------------------------------------------------------
extern "C" void kernel_launch(void* const* d_in, const int* in_sizes, int n_in,
                              void* d_out, int out_size, void* d_ws, size_t ws_size,
                              hipStream_t stream)
{
  (void)in_sizes; (void)n_in; (void)out_size; (void)ws_size;

  const float* x   = (const float*)d_in[0];
  const int*   pw1 = (const int*)  d_in[1];   // [4096][50]
  const int*   pw2 = (const int*)  d_in[2];   // [4096][256]
  const int*   pw3 = (const int*)  d_in[3];   // [10][256]
  const float* s1  = (const float*)d_in[4];
  const float* s2  = (const float*)d_in[5];
  const float* s3  = (const float*)d_in[6];
  const float* g1  = (const float*)d_in[7];
  const float* g2  = (const float*)d_in[8];
  float* out = (float*)d_out;                 // [16384][10]

  // workspace: bx [B,800]bf16 | h1 [B,4096]bf16 | h2 [B,4096]bf16 | partials
  char* ws = (char*)d_ws;
  u16* bx = (u16*)ws;
  u16* h1 = (u16*)(ws + (size_t)BATCH * IN_PAD * sizeof(u16));
  u16* h2 = h1 + (size_t)BATCH * HIDDEN;
  float* part = (float*)(h2 + (size_t)BATCH * HIDDEN);  // [KSPLIT][B][NPAD]

  const dim3 blk(256);
  const dim3 gemm_full((HIDDEN + BN - 1) / BN, BATCH / BM, 1);       // 16x128
  const dim3 gemm_out ((NCLASS + BN - 1) / BN, BATCH / BM, KSPLIT);  // 1x128x8

  convert_pad_kernel<<<BATCH, blk, 0, stream>>>(x, bx);

  // layer 1: [B,800]bf16 x W1^T -> h1 bf16
  bitnet_gemm_wmma<true><<<gemm_full, blk, 0, stream>>>(
      bx, IN_PAD, pw1, IN_F / 16 + 1, IN_PAD / BK, s1, HIDDEN, HIDDEN,
      nullptr, h1, HIDDEN, 0);
  rmsnorm_gelu_kernel<<<BATCH, blk, 0, stream>>>(h1, g1);

  // layer 2: [B,4096]bf16 x W2^T -> h2 bf16
  bitnet_gemm_wmma<true><<<gemm_full, blk, 0, stream>>>(
      h1, HIDDEN, pw2, HIDDEN / 16, HIDDEN / BK, s2, HIDDEN, HIDDEN,
      nullptr, h2, HIDDEN, 0);
  rmsnorm_gelu_kernel<<<BATCH, blk, 0, stream>>>(h2, g2);

  // layer 3: [B,4096]bf16 x W3^T, K split 8 ways -> f32 partials [8][B][16]
  bitnet_gemm_wmma<false><<<gemm_out, blk, 0, stream>>>(
      h2, HIDDEN, pw3, HIDDEN / 16, (HIDDEN / BK) / KSPLIT, s3, NCLASS, NPAD,
      part, nullptr, NPAD, (size_t)BATCH * NPAD);

  // deterministic partial reduction -> out [B][10]
  ksplit_reduce_kernel<<<(BATCH * NCLASS + 255) / 256, blk, 0, stream>>>(
      part, out);
}
